// CVAModule_712964571285
// MI455X (gfx1250) — matrix operand order repaired and checked by reading, not compiled, not run on hardware
//
#include <hip/hip_runtime.h>
#include <hip/hip_bf16.h>
#include <math.h>

// ---------------------------------------------------------------------------
// Problem constants (from reference)
// ---------------------------------------------------------------------------
constexpr int B      = 8;
constexpr int C      = 384;
constexpr int H      = 32;
constexpr int W      = 32;
constexpr int HEADS  = 12;
constexpr int GROUPS = 3;
constexpr int CG     = C / GROUPS;   // 128
constexpr int HD     = C / HEADS;    // 32
constexpr int NS     = H * W;        // 1024
constexpr float EPS  = 1e-5f;

typedef __attribute__((ext_vector_type(16))) _Float16 v16h;
typedef __attribute__((ext_vector_type(8)))  float    v8f;

#define WMMA_F16(a, b, c) \
    __builtin_amdgcn_wmma_f32_16x16x32_f16(false, (a), false, (b), (short)0, (c), false, false)

// Padded LDS row strides (halves / floats) to break bank-conflict patterns.
constexpr int LDH = 40;   // for 32-wide f16 tiles
constexpr int LDS_F = 33; // for 32-wide f32 tiles

// ---------------------------------------------------------------------------
// Kernel 1: batched 1x1-conv GEMM via WMMA.
//   Y[b,m,n] = sum_k Wm[m,k] * X[b,k,n] + bias[m]  (+ resid[b,m,n] if given)
// One wave computes a 64(M) x 16(N) tile: one shared B fragment per K-step,
// 4 A fragments loaded straight from global (K contiguous along W rows),
// 4 wmmas per K-step.  X tile staged to LDS *transposed* ([n][k]) so the
// B fragment is two contiguous 16B LDS reads per lane.
// Fragment layouts per CDNA5 ISA 7.12.2 (wave32):
//   A (16x32 f16): lane L -> row M=L&15; half h=L>>4 owns K in
//                  [8h..8h+7] and [8h+16..8h+23]
//   B (32x16 f16): lane L -> col N=L&15; half h owns K in [8h..8h+7] and [8h+16..8h+23]
//   C/D (16x16 f32): lane L -> col N=L&15; vgpr r -> row M = r + 8*(L>>4)
// ---------------------------------------------------------------------------
__global__ __launch_bounds__(32)
void gemm_wmma(const float* __restrict__ Wm, const float* __restrict__ X,
               const float* __restrict__ bias, const float* __restrict__ resid,
               float* __restrict__ Y, int M, int N, int K)
{
    __shared__ _Float16 bT[16][LDH];   // X tile, transposed: [n][k]

    const int l    = threadIdx.x;
    const int half = l >> 4;
    const int lo   = l & 15;
    const int n0   = blockIdx.x * 16;
    const int m0   = blockIdx.y * 64;
    const int b    = blockIdx.z;
    const int kb   = 8 * half;
    const int kh   = 16 * half;

    const float* Xb = X + (size_t)b * K * N;
    v8f acc[4] = {{}, {}, {}, {}};

    for (int k0 = 0; k0 < K; k0 += 32) {
        // --- stage X tile transposed: lane stages row c = k0+l, cols n0..n0+15
        {
            const float* src = Xb + (size_t)(k0 + l) * N + n0;
#pragma unroll
            for (int j = 0; j < 16; j += 4) {
                float4 f = *reinterpret_cast<const float4*>(src + j);
                bT[j + 0][l] = (_Float16)f.x;
                bT[j + 1][l] = (_Float16)f.y;
                bT[j + 2][l] = (_Float16)f.z;
                bT[j + 3][l] = (_Float16)f.w;
            }
        }
        if (k0 + 32 < K)
            __builtin_prefetch(Xb + (size_t)(k0 + 32 + l) * N + n0, 0, 1);
        __syncthreads();

        // --- B fragment: two contiguous 16B runs per lane (same k order as A)
        v16h bf;
#pragma unroll
        for (int j = 0; j < 16; ++j)
            bf[j] = bT[lo][kb + (j & 7) + ((j >> 3) << 4)];

        // --- 4 A fragments direct from global (K contiguous) + 4 wmmas
#pragma unroll
        for (int mt = 0; mt < 4; ++mt) {
            const float* wr = Wm + (size_t)(m0 + mt * 16 + lo) * K + k0;
            float4 f0 = *reinterpret_cast<const float4*>(wr + kb);
            float4 f1 = *reinterpret_cast<const float4*>(wr + kb + 4);
            float4 f2 = *reinterpret_cast<const float4*>(wr + kb + 16);
            float4 f3 = *reinterpret_cast<const float4*>(wr + kb + 20);
            v16h a;
            a[0]  = (_Float16)f0.x; a[1]  = (_Float16)f0.y;
            a[2]  = (_Float16)f0.z; a[3]  = (_Float16)f0.w;
            a[4]  = (_Float16)f1.x; a[5]  = (_Float16)f1.y;
            a[6]  = (_Float16)f1.z; a[7]  = (_Float16)f1.w;
            a[8]  = (_Float16)f2.x; a[9]  = (_Float16)f2.y;
            a[10] = (_Float16)f2.z; a[11] = (_Float16)f2.w;
            a[12] = (_Float16)f3.x; a[13] = (_Float16)f3.y;
            a[14] = (_Float16)f3.z; a[15] = (_Float16)f3.w;
            acc[mt] = WMMA_F16(a, bf, acc[mt]);
        }
        __syncthreads();
    }

    // --- epilogue: bias (+ residual), scatter per D layout
    const size_t yb = (size_t)b * M * N;
#pragma unroll
    for (int mt = 0; mt < 4; ++mt) {
#pragma unroll
        for (int r = 0; r < 8; ++r) {
            const int m = m0 + mt * 16 + r + 8 * half;
            const int n = n0 + lo;
            float v = acc[mt][r] + bias[m];
            if (resid) v += resid[yb + (size_t)m * N + n];
            Y[yb + (size_t)m * N + n] = v;
        }
    }
}

// ---------------------------------------------------------------------------
// Kernel 2: depthwise 5x5 conv -> channel LayerNorm -> GELU(tanh) ->
//           2xCG offset projection -> bilinear sample of x2 -> samp
// One 128-thread block per (group-batch bg, pixel n); thread = channel.
// ---------------------------------------------------------------------------
__device__ __forceinline__ float block_reduce_sum(float v, float* red, int c)
{
    red[c] = v;
    __syncthreads();
    for (int s = 64; s > 0; s >>= 1) {
        if (c < s) red[c] += red[c + s];
        __syncthreads();
    }
    float r = red[0];
    __syncthreads();
    return r;
}

__global__ __launch_bounds__(128)
void offset_sample(const float* __restrict__ qbuf, const float* __restrict__ x2,
                   const float* __restrict__ off_dw, const float* __restrict__ off_dw_b,
                   const float* __restrict__ ln_g, const float* __restrict__ ln_b,
                   const float* __restrict__ off_w, float* __restrict__ samp)
{
    __shared__ float red[128];
    __shared__ float fw[2];   // wx, wy
    __shared__ int   iw[4];   // x0, x1, y0, y1

    const int c  = threadIdx.x;       // channel within group
    const int n  = blockIdx.x;        // pixel
    const int bg = blockIdx.y;        // b*GROUPS + g
    const int y  = n / W;
    const int x  = n % W;

    // depthwise 5x5, SAME (zero pad)
    const float* qg = qbuf + ((size_t)bg * CG + c) * NS;
    float t = off_dw_b[c];
#pragma unroll
    for (int dy = 0; dy < 5; ++dy) {
        const int yy = y + dy - 2;
        if (yy < 0 || yy >= H) continue;
#pragma unroll
        for (int dx = 0; dx < 5; ++dx) {
            const int xx = x + dx - 2;
            if (xx < 0 || xx >= W) continue;
            t += qg[yy * W + xx] * off_dw[c * 25 + dy * 5 + dx];
        }
    }

    // LayerNorm over CG channels
    const float invCG = 1.0f / (float)CG;
    const float mu  = block_reduce_sum(t, red, c) * invCG;
    const float d   = t - mu;
    const float var = block_reduce_sum(d * d, red, c) * invCG;
    float tn = d * rsqrtf(var + EPS) * ln_g[c] + ln_b[c];

    // GELU (tanh approximation, jax.nn.gelu default)
    const float g = 0.5f * tn *
        (1.0f + tanhf(0.7978845608028654f * (tn + 0.044715f * tn * tn * tn)));

    // offset projection: off[o] = sum_c off_w[o,c] * g_c
    const float offx = block_reduce_sum(g * off_w[c],      red, c);
    const float offy = block_reduce_sum(g * off_w[CG + c], red, c);

    if (c == 0) {
        const float gx = ((x + 0.5f) / (float)W) * 2.0f - 1.0f;
        const float gy = ((y + 0.5f) / (float)H) * 2.0f - 1.0f;
        const float px = fminf(1.0f, fmaxf(-1.0f, offx + gx));
        const float py = fminf(1.0f, fmaxf(-1.0f, offy + gy));
        const float ix = (px + 1.0f) * 0.5f * (float)(W - 1);
        const float iy = (py + 1.0f) * 0.5f * (float)(H - 1);
        const float x0f = floorf(ix), y0f = floorf(iy);
        fw[0] = ix - x0f;
        fw[1] = iy - y0f;
        int x0 = min(W - 1, max(0, (int)x0f));
        int y0 = min(H - 1, max(0, (int)y0f));
        iw[0] = x0;  iw[1] = min(W - 1, x0 + 1);
        iw[2] = y0;  iw[3] = min(H - 1, y0 + 1);
    }
    __syncthreads();

    const float wx = fw[0], wy = fw[1];
    const int x0 = iw[0], x1i = iw[1], y0 = iw[2], y1i = iw[3];
    const float* xg = x2 + ((size_t)bg * CG + c) * NS;
    const float v00 = xg[y0  * W + x0 ];
    const float v01 = xg[y0  * W + x1i];
    const float v10 = xg[y1i * W + x0 ];
    const float v11 = xg[y1i * W + x1i];
    samp[((size_t)bg * CG + c) * NS + n] =
        v00 * (1.f - wx) * (1.f - wy) + v01 * wx * (1.f - wy) +
        v10 * (1.f - wx) * wy         + v11 * wx * wy;
}

// ---------------------------------------------------------------------------
// Kernel 3: flash attention via WMMA.  One wave per (b, head, 32-query tile).
// HD=32 matches the f16 WMMA K exactly.  Per 32-key chunk:
//   4 wmmas (2 Q-frag x 2 K-frag scores) + online softmax (all 32 lanes
//   own one of the 32 query rows) + 4 wmmas (P x V^T).
// K is staged into LDS transposed ([key][d]) and V natural ([d][key]) so
// every B fragment is two contiguous 16B LDS reads per lane.
// Q/K/V/O all laid out (B, C, NS) with channel = h*HD + d.
// ---------------------------------------------------------------------------
__global__ __launch_bounds__(32)
void attn_wmma(const float* __restrict__ Q, const float* __restrict__ Kb,
               const float* __restrict__ Vb, float* __restrict__ O)
{
    __shared__ _Float16 kT[32][LDH];   // [key][d]  (transposed)
    __shared__ _Float16 vT[32][LDH];   // [d][key]
    __shared__ float    sT[32][LDS_F]; // scores [q][key]
    __shared__ _Float16 pT[32][LDH];   // probs  [q][key]
    __shared__ float    alphaS[32];

    const int l    = threadIdx.x;
    const int half = l >> 4;
    const int lo   = l & 15;
    const int qt   = blockIdx.x;       // 32-query tile
    const int h    = blockIdx.y;
    const int b    = blockIdx.z;
    const int kb   = 8 * half;
    const int kh   = 16 * half;

    const size_t base = (size_t)b * C * NS + (size_t)h * HD * NS;
    const float scale = rsqrtf((float)HD);

    // Q^T fragments (A-layout, M=query, K=d), pre-scaled
    v16h aq0, aq1;
    const int qr0 = qt * 32 + lo;
    const int qr1 = qt * 32 + 16 + lo;
#pragma unroll
    for (int j = 0; j < 8; ++j) {
        aq0[j]     = (_Float16)(Q[base + (size_t)(kb + j)      * NS + qr0] * scale);
        aq0[8 + j] = (_Float16)(Q[base + (size_t)(kb + 16 + j) * NS + qr0] * scale);
        aq1[j]     = (_Float16)(Q[base + (size_t)(kb + j)      * NS + qr1] * scale);
        aq1[8 + j] = (_Float16)(Q[base + (size_t)(kb + 16 + j) * NS + qr1] * scale);
    }

    v8f o00 = {}, o01 = {}, o10 = {}, o11 = {};
    float m_i = -1e30f, l_i = 0.0f;    // lane l owns query row l of the tile

    for (int kc = 0; kc < NS; kc += 32) {
        // --- stage K (transposed scatter) and V (natural): lane owns d = l
        {
            const float* ks = Kb + base + (size_t)l * NS + kc;
            const float* vs = Vb + base + (size_t)l * NS + kc;
#pragma unroll
            for (int j = 0; j < 32; j += 4) {
                float4 fk = *reinterpret_cast<const float4*>(ks + j);
                kT[j + 0][l] = (_Float16)fk.x;
                kT[j + 1][l] = (_Float16)fk.y;
                kT[j + 2][l] = (_Float16)fk.z;
                kT[j + 3][l] = (_Float16)fk.w;
                float4 fv = *reinterpret_cast<const float4*>(vs + j);
                vT[l][j + 0] = (_Float16)fv.x;  vT[l][j + 1] = (_Float16)fv.y;
                vT[l][j + 2] = (_Float16)fv.z;  vT[l][j + 3] = (_Float16)fv.w;
            }
        }
        __syncthreads();

        // --- score fragments (contiguous LDS reads) + 4 wmmas
        v16h b0, b1;
#pragma unroll
        for (int j = 0; j < 16; ++j) {
            b0[j] = kT[lo][kb + (j & 7) + ((j >> 3) << 4)];        // key = kc + lo
            b1[j] = kT[16 + lo][kb + (j & 7) + ((j >> 3) << 4)];   // key = kc + 16 + lo
        }
        v8f s00 = {}, s01 = {}, s10 = {}, s11 = {};
        s00 = WMMA_F16(aq0, b0, s00);
        s01 = WMMA_F16(aq0, b1, s01);
        s10 = WMMA_F16(aq1, b0, s10);
        s11 = WMMA_F16(aq1, b1, s11);

        // --- spill scores (C-layout) to LDS
#pragma unroll
        for (int r = 0; r < 8; ++r) {
            sT[r + 8 * half][lo]           = s00[r];
            sT[r + 8 * half][16 + lo]      = s01[r];
            sT[16 + r + 8 * half][lo]      = s10[r];
            sT[16 + r + 8 * half][16 + lo] = s11[r];
        }
        __syncthreads();

        // --- online softmax: every lane owns one of the 32 query rows
        {
            float mx = m_i;
#pragma unroll
            for (int j = 0; j < 32; ++j) mx = fmaxf(mx, sT[l][j]);
            const float alpha = __expf(m_i - mx);
            float sum = 0.0f;
#pragma unroll
            for (int j = 0; j < 32; ++j) {
                const float p = __expf(sT[l][j] - mx);
                pT[l][j] = (_Float16)p;
                sum += p;
            }
            l_i = l_i * alpha + sum;
            m_i = mx;
            alphaS[l] = alpha;
        }
        __syncthreads();

        // --- rescale running output
#pragma unroll
        for (int r = 0; r < 8; ++r) {
            const float a0 = alphaS[r + 8 * half];
            const float a1 = alphaS[16 + r + 8 * half];
            o00[r] *= a0;  o01[r] *= a0;
            o10[r] *= a1;  o11[r] *= a1;
        }

        // --- P fragments (A-layout) and V^T fragments (B-layout, N=d, K=key)
        v16h pa0, pa1, vb0, vb1;
#pragma unroll
        for (int j = 0; j < 8; ++j) {
            pa0[j]     = pT[lo][kb + j];
            pa0[8 + j] = pT[lo][kb + 16 + j];
            pa1[j]     = pT[16 + lo][kb + j];
            pa1[8 + j] = pT[16 + lo][kb + 16 + j];
        }
#pragma unroll
        for (int j = 0; j < 16; ++j) {
            vb0[j] = vT[lo][kb + (j & 7) + ((j >> 3) << 4)];        // d = lo
            vb1[j] = vT[16 + lo][kb + (j & 7) + ((j >> 3) << 4)];   // d = 16 + lo
        }
        o00 = WMMA_F16(pa0, vb0, o00);
        o01 = WMMA_F16(pa0, vb1, o01);
        o10 = WMMA_F16(pa1, vb0, o10);
        o11 = WMMA_F16(pa1, vb1, o11);
        __syncthreads();
    }

    // --- finalize: divide by row sums, store O[b, h*HD+d, qt*32+q]
    alphaS[l] = 1.0f / l_i;
    __syncthreads();
#pragma unroll
    for (int r = 0; r < 8; ++r) {
        const int q0 = r + 8 * half;
        const int q1 = 16 + q0;
        const float i0 = alphaS[q0];
        const float i1 = alphaS[q1];
        const int qi0 = qt * 32 + q0;
        const int qi1 = qt * 32 + q1;
        O[base + (size_t)(lo)      * NS + qi0] = o00[r] * i0;
        O[base + (size_t)(16 + lo) * NS + qi0] = o01[r] * i0;
        O[base + (size_t)(lo)      * NS + qi1] = o10[r] * i1;
        O[base + (size_t)(16 + lo) * NS + qi1] = o11[r] * i1;
    }
}

// ---------------------------------------------------------------------------
// Launcher. Inputs (setup_inputs order):
//  0:x1 1:x2 2:Wq 3:bq 4:Wk 5:bk 6:Wv 7:bv 8:Wo 9:bo
// 10:off_dw 11:off_dw_b 12:ln_g 13:ln_b 14:off_w
// ---------------------------------------------------------------------------
extern "C" void kernel_launch(void* const* d_in, const int* in_sizes, int n_in,
                              void* d_out, int out_size, void* d_ws, size_t ws_size,
                              hipStream_t stream)
{
    const float* x1       = (const float*)d_in[0];
    const float* x2       = (const float*)d_in[1];
    const float* Wq       = (const float*)d_in[2];
    const float* bq       = (const float*)d_in[3];
    const float* Wk       = (const float*)d_in[4];
    const float* bk       = (const float*)d_in[5];
    const float* Wv       = (const float*)d_in[6];
    const float* bv       = (const float*)d_in[7];
    const float* Wo       = (const float*)d_in[8];
    const float* bo       = (const float*)d_in[9];
    const float* off_dw   = (const float*)d_in[10];
    const float* off_dw_b = (const float*)d_in[11];
    const float* ln_g     = (const float*)d_in[12];
    const float* ln_b     = (const float*)d_in[13];
    const float* off_w    = (const float*)d_in[14];

    const size_t T = (size_t)B * C * NS;    // 3,145,728 floats per tensor
    float* ws     = (float*)d_ws;
    float* qws    = ws;            // q projection      (B,C,NS)
    float* sampws = ws + 1 * T;    // deformable sample (B,C,NS)
    float* kws    = ws + 2 * T;    // k projection
    float* vws    = ws + 3 * T;    // v projection
    float* aows   = ws + 4 * T;    // attention output

    const dim3 gGemm(NS / 16, C / 64, B);   // (64, 6, 8), 1 wave each, 64x16 tile

    // q = Wq x1 + bq
    gemm_wmma<<<gGemm, 32, 0, stream>>>(Wq, x1, bq, nullptr, qws, C, NS, C);

    // offsets + bilinear sampling of x2
    offset_sample<<<dim3(NS, B * GROUPS), 128, 0, stream>>>(
        qws, x2, off_dw, off_dw_b, ln_g, ln_b, off_w, sampws);

    // k = Wk samp + bk ; v = Wv samp + bv
    gemm_wmma<<<gGemm, 32, 0, stream>>>(Wk, sampws, bk, nullptr, kws, C, NS, C);
    gemm_wmma<<<gGemm, 32, 0, stream>>>(Wv, sampws, bv, nullptr, vws, C, NS, C);

    // attention: one wave per (b, head, 32-query tile)
    attn_wmma<<<dim3(NS / 32, HEADS, B), 32, 0, stream>>>(qws, kws, vws, aows);

    // y = x1 + Wo attn_out + bo
    gemm_wmma<<<gGemm, 32, 0, stream>>>(Wo, aows, bo, x1, (float*)d_out, C, NS, C);
}
